// RouteLayer_43319040147907
// MI455X (gfx1250) — compile-verified, matrix operand order checked by reading, not run on hardware
//
#include <hip/hip_runtime.h>
#include <hip/hip_bf16.h>

#define B_ 64
#define I_ 1152
#define J_ 64
#define D_ 32
#define K_ 64
#define M_ 2048   // J_*D_

typedef __attribute__((ext_vector_type(16))) __bf16 v16bf;
typedef __attribute__((ext_vector_type(8)))  float  v8f;

union BFV { v16bf v; unsigned u32[8]; unsigned short u[16]; };

// Pack two f32 into packed bf16 (truncation) with a single v_perm_b32.
__device__ __forceinline__ unsigned pack_bf16(float lo, float hi) {
    union { float f; unsigned u; } a, b; a.f = hi; b.f = lo;
    // result bytes: [1:0] = lo.bytes[3:2], [3:2] = hi.bytes[3:2]
    return __builtin_amdgcn_perm(a.u, b.u, 0x07060302u);
}
__device__ __forceinline__ unsigned short bf_trunc(float f) {
    union { float f; unsigned u; } x; x.f = f;
    return (unsigned short)(x.u >> 16);        // folds into ds_store_b16_d16_hi
}
__device__ __forceinline__ unsigned short f2bf_rne(float f) {
    union { float f; unsigned u; } x; x.f = f;
    unsigned r = x.u + 0x7FFFu + ((x.u >> 16) & 1u);
    return (unsigned short)(r >> 16);
}
__device__ __forceinline__ float bf2f(unsigned short h) {
    union { unsigned u; float f; } x; x.u = ((unsigned)h) << 16;
    return x.f;
}

__global__ void zero_kernel(float* p) {
    int idx = blockIdx.x * 256 + threadIdx.x;
    if (idx < M_ * B_) p[idx] = 0.0f;
}

// One block = one capsule i  x  one half of B (32 batch cols).
// T = routing iteration (0,1,2).
template <int T>
__global__ __launch_bounds__(256)
void route_kernel(const float* __restrict__ x, const float* __restrict__ W,
                  const float* __restrict__ vT, float* __restrict__ bij,
                  float* __restrict__ sT) {
    const int i    = blockIdx.x;          // 0..1151
    const int gy   = blockIdx.y;          // 0..1  (b half)
    const int tid  = threadIdx.x;         // 0..255
    const int wave = tid >> 5;            // 0..7
    const int lane = tid & 31;
    const int l16  = lane & 15;
    const int kb   = (lane < 16) ? 0 : 8; // k-subset per A/B 16-bit layout

    __shared__ unsigned short uSh[M_ * 32];              // u_hat bf16 [m][bl] 128 KB
    __shared__ __align__(32) unsigned short xOp[2][2][32][16]; // B frags [kh][nt][lane][e] 4 KB
    __shared__ float ab[32 * J_];                        // a / c  [bl][j]       8 KB

    // ---- stage x directly into B-operand lane layout (bf16) ----
    for (int idx = tid; idx < K_ * 32; idx += 256) {
        int k = idx & 63, bl = idx >> 6;
        float v = x[((size_t)(gy * 32 + bl) * I_ + i) * K_ + k];
        int kh = k >> 5, kr = k & 31;
        int lhalf = (kr >> 3) & 1;                  // k in [8,16)|[24,32) -> upper lanes
        int e = (kr & 7) + ((kr >> 4) << 3);        // element slot 0..15
        xOp[kh][bl >> 4][lhalf * 16 + (bl & 15)][e] = f2bf_rne(v);
    }
    if (T > 0)
        for (int idx = tid; idx < 32 * J_; idx += 256) ab[idx] = 0.0f;
    __syncthreads();

    // ---- preload B operands: one 32B vector ds read per fragment ----
    BFV bop[2][2];
#pragma unroll
    for (int nt = 0; nt < 2; ++nt)
#pragma unroll
        for (int kh = 0; kh < 2; ++kh)
            bop[nt][kh].v = *(const v16bf*)&xOp[kh][nt][lane][0];

    // ---- GEMM over m-tiles; epilogue stores u_hat(bf16) + agreement partials ----
    for (int q = 0; q < 16; ++q) {
        const int mt   = wave * 16 + q;          // 0..127
        const int mrow = mt * 16 + l16;          // this lane's A row (m = j*32+d)
        const float4* w4 = (const float4*)(W + ((size_t)i * M_ + mrow) * K_ + kb);
        // k runs: [kb, kb+8), [kb+16, kb+24), (+32 for second fragment)
        float4 p0 = w4[0],  p1 = w4[1],  p2 = w4[4],  p3 = w4[5];
        float4 q0 = w4[8],  q1 = w4[9],  q2 = w4[12], q3 = w4[13];
        BFV a0, a1;
        a0.u32[0] = pack_bf16(p0.x, p0.y); a0.u32[1] = pack_bf16(p0.z, p0.w);
        a0.u32[2] = pack_bf16(p1.x, p1.y); a0.u32[3] = pack_bf16(p1.z, p1.w);
        a0.u32[4] = pack_bf16(p2.x, p2.y); a0.u32[5] = pack_bf16(p2.z, p2.w);
        a0.u32[6] = pack_bf16(p3.x, p3.y); a0.u32[7] = pack_bf16(p3.z, p3.w);
        a1.u32[0] = pack_bf16(q0.x, q0.y); a1.u32[1] = pack_bf16(q0.z, q0.w);
        a1.u32[2] = pack_bf16(q1.x, q1.y); a1.u32[3] = pack_bf16(q1.z, q1.w);
        a1.u32[4] = pack_bf16(q2.x, q2.y); a1.u32[5] = pack_bf16(q2.z, q2.w);
        a1.u32[6] = pack_bf16(q3.x, q3.y); a1.u32[7] = pack_bf16(q3.z, q3.w);

        const int j     = mt >> 1;                     // D=32 -> one j per m-tile
        const int mbase = mt * 16 + ((lane < 16) ? 0 : 8);
#pragma unroll
        for (int nt = 0; nt < 2; ++nt) {
            v8f acc = {};
            acc = __builtin_amdgcn_wmma_f32_16x16x32_bf16(
                false, a0.v, false, bop[nt][0].v, (short)0, acc, false, false);
            acc = __builtin_amdgcn_wmma_f32_16x16x32_bf16(
                false, a1.v, false, bop[nt][1].v, (short)0, acc, false, false);
            const int bl = nt * 16 + l16;              // local batch col
#pragma unroll
            for (int r = 0; r < 8; ++r)
                uSh[(mbase + r) * 32 + bl] = bf_trunc(acc[r]);
            if (T > 0) {
                const int bg = gy * 32 + bl;
                float part = 0.0f;
#pragma unroll
                for (int r = 0; r < 8; ++r)
                    part += acc[r] * vT[(mbase + r) * B_ + bg];
                atomicAdd(&ab[bl * J_ + j], part);     // ds_add_f32
            }
        }
    }
    __syncthreads();

    // ---- b update + softmax over j (wave handles 4 rows, lane = j and j+32) ----
    if (T > 0) {
        for (int rr = 0; rr < 4; ++rr) {
            const int bl = wave * 4 + rr;
            const int bg = gy * 32 + bl;
            float v0 = ab[bl * J_ + lane];
            float v1 = ab[bl * J_ + 32 + lane];
            if (T == 2) {
                v0 += bij[i * 4096 + bg * J_ + lane];
                v1 += bij[i * 4096 + bg * J_ + 32 + lane];
            }
            if (T == 1) {  // persist b for next iteration
                bij[i * 4096 + bg * J_ + lane]      = v0;
                bij[i * 4096 + bg * J_ + 32 + lane] = v1;
            }
            float mx = fmaxf(v0, v1);
            for (int o = 16; o; o >>= 1) mx = fmaxf(mx, __shfl_xor(mx, o, 32));
            float e0 = __expf(v0 - mx), e1 = __expf(v1 - mx);
            float sm = e0 + e1;
            for (int o = 16; o; o >>= 1) sm += __shfl_xor(sm, o, 32);
            float inv = 1.0f / sm;
            ab[bl * J_ + lane]      = e0 * inv;
            ab[bl * J_ + 32 + lane] = e1 * inv;
        }
    }
    __syncthreads();

    // ---- s[b,j,d] += c[b,j] * u_hat : global f32 atomics into sT[m][b] ----
    const float cu = 1.0f / (float)J_;
    for (int idx = tid; idx < M_ * 32; idx += 256) {
        const int bl = idx & 31, m = idx >> 5;
        float uv = bf2f(uSh[m * 32 + bl]);
        float c  = (T == 0) ? cu : ab[bl * J_ + (m >> 5)];
        unsafeAtomicAdd(&sT[m * B_ + gy * 32 + bl], uv * c);  // global_atomic_add_f32
    }
}

// One wave per (b,j); lane = d (D=32 < wave32=32 lanes). FINAL writes (b,j,d)
// output, otherwise writes vT in [(j*32+d)][b] layout for the next iteration.
template <bool FINAL>
__global__ __launch_bounds__(256)
void squash_kernel(const float* __restrict__ sT, float* __restrict__ out) {
    int gid  = blockIdx.x * 256 + threadIdx.x;
    int wid  = gid >> 5;                 // 0..4095
    int lane = threadIdx.x & 31;         // d
    int b = wid >> 6, j = wid & 63;
    float val = sT[(j * D_ + lane) * B_ + b];
    float n2 = val * val;
    for (int o = 16; o; o >>= 1) n2 += __shfl_xor(n2, o, 32);
    float scale = n2 / (1.0f + n2) * rsqrtf(n2 + 1e-9f);
    float r = val * scale;
    if (FINAL) out[(b * J_ + j) * D_ + lane] = r;
    else       out[(j * D_ + lane) * B_ + b] = r;
}

extern "C" void kernel_launch(void* const* d_in, const int* in_sizes, int n_in,
                              void* d_out, int out_size, void* d_ws, size_t ws_size,
                              hipStream_t stream) {
    (void)in_sizes; (void)n_in; (void)out_size; (void)ws_size;
    const float* x = (const float*)d_in[0];
    const float* W = (const float*)d_in[1];
    float* out = (float*)d_out;
    float* ws  = (float*)d_ws;
    float* sT  = ws;                 // 131072 f32
    float* vT  = ws + 131072;        // 131072 f32
    float* bij = ws + 262144;        // 4718592 f32  (~19.9 MB total)

    dim3 rgrid(I_, 2), rblk(256);

    zero_kernel<<<512, 256, 0, stream>>>(sT);
    route_kernel<0><<<rgrid, rblk, 0, stream>>>(x, W, vT, bij, sT);
    squash_kernel<false><<<512, 256, 0, stream>>>(sT, vT);

    zero_kernel<<<512, 256, 0, stream>>>(sT);
    route_kernel<1><<<rgrid, rblk, 0, stream>>>(x, W, vT, bij, sT);
    squash_kernel<false><<<512, 256, 0, stream>>>(sT, vT);

    zero_kernel<<<512, 256, 0, stream>>>(sT);
    route_kernel<2><<<rgrid, rblk, 0, stream>>>(x, W, vT, bij, sT);
    squash_kernel<true><<<512, 256, 0, stream>>>(sT, out);
}
